// LSTMModel_74543452389574
// MI455X (gfx1250) — matrix operand order repaired and checked by reading, not compiled, not run on hardware
//
#include <hip/hip_runtime.h>
#include <stdint.h>

typedef __bf16 bf16_t;
typedef bf16_t v16bf __attribute__((ext_vector_type(16)));
typedef float  v8f   __attribute__((ext_vector_type(8)));
typedef unsigned int   u32;
typedef unsigned short u16;

#define T_STEPS 1000
#define NT      13                      // ceil(200/16) gate column tiles
#define B_SLOT        48                // bytes per lane per B fragment slot (32B data + pad)
#define B_SLICE_BYTES (32 * B_SLOT)     // 1536 per (tile,kslice)
#define B_REGION      (NT * 2 * B_SLICE_BYTES)   // 39936
#define STG_STRIDE    144               // staging row stride bytes (72 bf16, 16B aligned, bank-spread)
#define STG_BYTES     (16 * STG_STRIDE) // 2304 (one 16-row tile per workgroup)
#define GT_STRIDE     (208 * 4)         // gate row stride bytes (f32)
#define GT_BYTES      (16 * GT_STRIDE)  // 13312
#define SMEM_TOTAL    (B_REGION + STG_BYTES + GT_BYTES)   // 55552

__device__ __forceinline__ u16 f2bf(float f) {
  u32 u = __builtin_bit_cast(u32, f);
  u += 0x7FFFu + ((u >> 16) & 1u);      // round-to-nearest-even
  return (u16)(u >> 16);
}
__device__ __forceinline__ float sigm(float x) {
  return __builtin_amdgcn_rcpf(1.0f + __expf(-x));
}
__device__ __forceinline__ float tanh_fast(float x) {
  float e = __expf(-2.0f * x);
  return 1.0f - 2.0f * __builtin_amdgcn_rcpf(1.0f + e);
}

union FragU { uint4 u[2]; v16bf v; };
__device__ __forceinline__ v16bf ldfrag(const char* p) {
  FragU f;
  f.u[0] = *(const uint4*)(p);
  f.u[1] = *(const uint4*)(p + 16);
  return f.v;
}

__global__ __launch_bounds__(128)
void lstm_wmma_kernel(const float* __restrict__ x,
                      const float* __restrict__ W_ih,
                      const float* __restrict__ W_hh,
                      const float* __restrict__ b_ih,
                      const float* __restrict__ b_hh,
                      const float* __restrict__ W_lin,
                      const float* __restrict__ b_lin,
                      float* __restrict__ out) {
  extern __shared__ char smem[];
  const int tid  = threadIdx.x;
  const int lane = tid & 31;
  // Scalar (SGPR) wave id so uniform guards compile to s_cbranch, not EXEC
  // masking: WMMA must never execute with EXEC != all-ones.
  const int wave = __builtin_amdgcn_readfirstlane(tid >> 5);
  const int m16  = lane & 15;           // A-row / C-column within tile
  const int kh   = lane >> 4;           // lane half (K-split in A, M-split in C)
  const int rowBase = blockIdx.x * 16;  // one 16-row batch tile per workgroup

  const u32 stgOff = (u32)B_REGION;
  const u32 gtOff  = (u32)(B_REGION + STG_BYTES);

  // ---- build augmented B matrix [W_hh.T ; W_ih.T ; 0-pad] (64 x 208, bf16),
  //      stored in WMMA B-fragment order: slice = nt*2+ks, 48B lane slots.
  for (int idx = tid; idx < NT * 2 * 32 * 16; idx += 128) {
    int hidx  = idx & 15;
    int ln    = (idx >> 4) & 31;
    int sl    = idx >> 9;               // nt*2 + ks
    int ks    = sl & 1;
    int nt    = sl >> 1;
    int n     = ln & 15;
    int khalf = ln >> 4;
    int k     = ks * 32 + khalf * 16 + hidx;
    int g     = nt * 16 + n;
    float v = 0.0f;
    if (g < 200) {
      if (k < 50)       v = W_hh[g * 50 + k];
      else if (k == 50) v = W_ih[g * 2 + 0];
      else if (k == 51) v = W_ih[g * 2 + 1];
    }
    *(u16*)(smem + (u32)(sl * B_SLICE_BYTES + ln * B_SLOT + hidx * 2)) = f2bf(v);
  }

  // ---- zero staging area (h0 = 0, x slots, K pad): 576 dwords
  for (int idx = tid; idx < 576; idx += 128)
    *(u32*)(smem + stgOff + (u32)(idx * 4)) = 0u;

  __syncthreads();

  // ---- this wave's N-tiles (4/3/3/3 split) -> B fragments live in VGPRs for all T
  const int ntBase = (wave == 0) ? 0 : (1 + 3 * wave);   // 0,4,7,10
  const int ntCnt  = (wave == 0) ? 4 : 3;
  v16bf bf0[4], bf1[4];
  float biasv[4];
  #pragma unroll
  for (int i = 0; i < 4; ++i) {
    biasv[i] = 0.0f;
    if (i < ntCnt) {
      int nt = ntBase + i;
      const char* bbase = smem + (u32)((nt * 2) * B_SLICE_BYTES + lane * B_SLOT);
      bf0[i] = ldfrag(bbase);
      bf1[i] = ldfrag(bbase + B_SLICE_BYTES);
      int g = nt * 16 + m16;
      if (g < 200) biasv[i] = b_ih[g] + b_hh[g];
    }
  }

  // ---- per-lane ownership of up to 7 (m,j) cells: e = tid + 128*q, e < 800
  u32 gaddr[7]; u32 haddr[7]; float creg[7];
  #pragma unroll
  for (int q = 0; q < 7; ++q) {
    int e = tid + 128 * q;
    if (e > 799) e = 799;               // dummy (guarded at use)
    int m = e / 50;
    int j = e - 50 * m;
    gaddr[q] = gtOff + (u32)((m * 208 + j) * 4);
    haddr[q] = stgOff + (u32)(m * STG_STRIDE + j * 2);
    creg[q]  = 0.0f;
  }

  const float* xptr  = x + ((size_t)(rowBase + (lane >> 1)) * T_STEPS) * 2 + (lane & 1);
  char*       xslot  = smem + stgOff + (u32)((lane >> 1) * STG_STRIDE + (50 + (lane & 1)) * 2);
  const char* arow   = smem + stgOff + (u32)(m16 * STG_STRIDE + kh * 32);
  const u32   colOff = (u32)(m16 * 4);

  #pragma unroll 1
  for (int t = 0; t < T_STEPS; ++t) {
    // x_t into staging columns 50,51 (wave 0 only; disjoint from h columns)
    if (wave == 0) {
      *(u16*)xslot = f2bf(xptr[2 * t]);
      __builtin_prefetch(xptr + 2 * (t + 1), 0, 3);
    }
    __syncthreads();   // h_{t} (prev consume) + x_t visible to all waves

    // A fragments for row m16: K 0..31 and 32..63 (same A in every wave)
    v16bf a0, a1;
    {
      FragU f0, f1;
      f0.u[0] = *(const uint4*)(arow + 0);
      f0.u[1] = *(const uint4*)(arow + 16);
      f1.u[0] = *(const uint4*)(arow + 64);
      f1.u[1] = *(const uint4*)(arow + 80);
      a0 = f0.v; a1 = f1.v;
    }

    // gates tiles for this wave; distinct accumulators per tile so the
    // 8 WMMAs pipeline (no D-register WAW serialization); uniform scalar
    // branch skips tile 3 on waves 1-3 with EXEC untouched.
    v8f acc[4];
    #pragma unroll
    for (int i = 0; i < 4; ++i)
      if (i < ntCnt) {
        v8f z = {};
        acc[i] = __builtin_amdgcn_wmma_f32_16x16x32_bf16(false, a0, false, bf0[i], (short)0, z, false, false);
      }
    #pragma unroll
    for (int i = 0; i < 4; ++i)
      if (i < ntCnt)
        acc[i] = __builtin_amdgcn_wmma_f32_16x16x32_bf16(false, a1, false, bf1[i], (short)0, acc[i], false, false);
    #pragma unroll
    for (int i = 0; i < 4; ++i)
      if (i < ntCnt) {
        u32 cOff = gtOff + (u32)((ntBase + i) * 64) + colOff;
        #pragma unroll
        for (int r = 0; r < 8; ++r)
          *(float*)(smem + cOff + (u32)((r + 8 * kh) * GT_STRIDE)) = acc[i][r] + biasv[i];
      }
    __syncthreads();   // full 16x200 gate buffer ready

    // elementwise LSTM cell update (PyTorch gate order i,f,g,o); c lives in VGPRs
    const bool last = (t == T_STEPS - 1);
    #pragma unroll
    for (int q = 0; q < 7; ++q) {
      if (q < 6 || wave == 0) {         // e = tid + 128*q < 800 (tail is wave-uniform)
        float gi = *(const float*)(smem + gaddr[q]);
        float gf = *(const float*)(smem + gaddr[q] + 200);
        float gg = *(const float*)(smem + gaddr[q] + 400);
        float go = *(const float*)(smem + gaddr[q] + 600);
        gi = sigm(gi); gf = sigm(gf); gg = tanh_fast(gg); go = sigm(go);
        float c = gf * creg[q] + gi * gg;
        creg[q] = c;
        float h = go * tanh_fast(c);
        *(u16*)(smem + haddr[q]) = f2bf(h);        // h_{t+1} (bf16) -> A staging
        if (last) *(float*)(smem + gaddr[q]) = h;  // stash final h (f32) for the head
      }
    }
  }
  __syncthreads();

  // ---- linear head: out[b] = h_T . W_lin + b_lin   (one row per lane 0..15)
  if (tid < 16) {
    float acc = b_lin[0];
    for (int j = 0; j < 50; ++j)
      acc += *(const float*)(smem + gtOff + (u32)(tid * GT_STRIDE + j * 4)) * W_lin[j];
    out[rowBase + tid] = acc;
  }
}

extern "C" void kernel_launch(void* const* d_in, const int* in_sizes, int n_in,
                              void* d_out, int out_size, void* d_ws, size_t ws_size,
                              hipStream_t stream) {
  (void)in_sizes; (void)n_in; (void)out_size; (void)d_ws; (void)ws_size;
  const float* x     = (const float*)d_in[0];
  const float* W_ih  = (const float*)d_in[1];
  const float* W_hh  = (const float*)d_in[2];
  const float* b_ih  = (const float*)d_in[3];
  const float* b_hh  = (const float*)d_in[4];
  const float* W_lin = (const float*)d_in[5];
  const float* b_lin = (const float*)d_in[6];
  float* out = (float*)d_out;

  dim3 grid(4096 / 16);   // 256 workgroups x 4 waves; one 16-row WMMA tile each
  dim3 block(128);
  lstm_wmma_kernel<<<grid, block, SMEM_TOTAL, stream>>>(
      x, W_ih, W_hh, b_ih, b_hh, W_lin, b_lin, out);
}
